// MLXSwiGLU_5214090297425
// MI455X (gfx1250) — matrix operand order repaired and checked by reading, not compiled, not run on hardware
//
#include <hip/hip_runtime.h>
#include <hip/hip_bf16.h>
#include <stdint.h>

// ---------------------------------------------------------------------------
// SwiGLU MLP: out = (silu(x @ Wg^T) * (x @ Wu^T)) @ Wd^T
// bf16 WMMA GEMMs, f32 accumulation. LDS double-buffered K-pipeline fed by the
// CDNA5 Tensor Data Mover (tensor_load_to_lds + s_wait_tensorcnt). Wave tile
// 32x64; KC=64 stages (32 WMMAs between barriers); two-deep B-fragment
// register rotation keeps ds_loads ahead of the WMMA pipe.
// ---------------------------------------------------------------------------

#define DIM_IN  2048
#define DIM_HID 8192
#define NTOK    8192   // 4 * 2048 tokens

#define KC 64          // K-slice per pipeline stage
#define RS 72          // padded LDS row stride in bf16 elems (144 B = 36 banks)

typedef __attribute__((ext_vector_type(16))) __bf16 v16bf;
typedef __attribute__((ext_vector_type(8)))  __bf16 v8bf;
typedef __attribute__((ext_vector_type(4)))  __bf16 v4bf;
typedef __attribute__((ext_vector_type(8)))  float  v8f;
typedef __attribute__((ext_vector_type(4)))  float  v4f;
typedef __attribute__((ext_vector_type(4)))  unsigned int u32x4;
typedef __attribute__((ext_vector_type(8)))  int    i32x8;
typedef __attribute__((ext_vector_type(4)))  int    i32x4;

#if __has_builtin(__builtin_amdgcn_tensor_load_to_lds) && __has_builtin(__builtin_amdgcn_s_wait_tensorcnt)
#define USE_TDM 1
#else
#define USE_TDM 0
#endif

// A-matrix 16x32 bf16 fragment: lane<16 holds K 0..15 contiguous,
// lane>=16 holds K 16..31; p already includes the hi*16 element offset.
__device__ __forceinline__ v16bf load_frag_a(const __bf16* p) {
  v8bf lo = *(const v8bf*)(p);
  v8bf hi = *(const v8bf*)(p + 8);
  return __builtin_shufflevector(lo, hi, 0,1,2,3,4,5,6,7,8,9,10,11,12,13,14,15);
}

// B-matrix 32x16 bf16 fragment: lane<16 holds K 0..15 contiguous, lane>=16
// holds K 16..31; p already includes the hi*16 element offset.
__device__ __forceinline__ v16bf load_frag_b(const __bf16* p) {
  v8bf lo = *(const v8bf*)(p);
  v8bf hi = *(const v8bf*)(p + 8);
  return __builtin_shufflevector(lo, hi, 0,1,2,3,4,5,6,7,8,9,10,11,12,13,14,15);
}

__device__ __forceinline__ v8f wmma_bf16(v16bf a, v16bf b, v8f c) {
  return __builtin_amdgcn_wmma_f32_16x16x32_bf16(false, a, false, b, (short)0, c,
                                                 false, false);
}

#if USE_TDM
// TDM 2D-tile load: rows x KC bf16 from a row-major matrix (row stride =
// stride0 elements) into LDS at lptr. Row = KC*2 = 128 B = 32 DWORDs, so
// pad_interval=4 (32 DWORDs) + pad_amount=3 (4 DWORDs = 16 B) give a 144 B
// LDS row pitch (= RS elements). D# layout per CDNA5 ISA §8.
// 6-arg builtin: (u32x4 g0, i32x8 g1, i32x4 g2, i32x4 g3, i32x8 extra, cpol);
// groups 2/3 (3D+/gather) unused for 2D -> zero-filled.
__device__ __forceinline__ void tdm_load_2d(const void* gptr, const void* lptr,
                                            unsigned rows,
                                            unsigned long long stride0_elems) {
  unsigned long long ga = (unsigned long long)(uintptr_t)gptr;
  unsigned lds = (unsigned)(uintptr_t)lptr;   // low 32 bits = LDS offset
  u32x4 g0;
  g0[0] = 1u;                                     // count=1, user mode
  g0[1] = lds;                                    // lds_addr
  g0[2] = (unsigned)ga;                           // global_addr[31:0]
  g0[3] = ((unsigned)(ga >> 32) & 0x01FFFFFFu)    // global_addr[56:32]
          | 0x80000000u;                          // type=2 ("image")
  i32x8 g1;
  g1[0] = (int)((1u << 16)        // data_size = 1 -> 2 bytes
              | (1u << 20)        // pad_enable
              | (4u << 22)        // pad_interval: 32 DWORDs (128B) between pads
              | (3u << 25));      // pad_amount: 4 DWORDs (16B)
  g1[1] = (int)(((unsigned)KC & 0xFFFFu) << 16);  // tensor_dim0[15:0] = KC
  g1[2] = (int)((rows & 0xFFFFu) << 16);          // dim0 hi=0 | tensor_dim1 lo
  g1[3] = (int)((unsigned)KC << 16);              // dim1 hi=0 | tile_dim0 = KC
  g1[4] = (int)(rows & 0xFFFFu);                  // tile_dim1 = rows, tile_dim2=0
  g1[5] = (int)(unsigned)(stride0_elems & 0xFFFFFFFFull);   // stride0 lo
  g1[6] = (int)(unsigned)((stride0_elems >> 32) & 0xFFFFull); // stride0 hi
  g1[7] = 0;
  i32x4 gz4 = {0, 0, 0, 0};
  i32x8 gz8 = {0, 0, 0, 0, 0, 0, 0, 0};
  __builtin_amdgcn_tensor_load_to_lds(g0, g1, gz4, gz4, gz8, 0);
}
#endif

// ---------------------------------------------------------------------------
// fp32 -> bf16 conversion
// ---------------------------------------------------------------------------
__global__ void cvt_f32_to_bf16(const float* __restrict__ src,
                                __bf16* __restrict__ dst, int n4) {
  int i = blockIdx.x * blockDim.x + threadIdx.x;
  if (i < n4) {
    v4f v = ((const v4f*)src)[i];
    v4bf o;
    o.x = (__bf16)v.x; o.y = (__bf16)v.y; o.z = (__bf16)v.z; o.w = (__bf16)v.w;
    ((v4bf*)dst)[i] = o;
  }
}

// ---------------------------------------------------------------------------
// Fused gate+up GEMM + SiLU epilogue.
// Block 256 thr = 8 waves (4M x 2N). Block tile 128(M) x 128(N); wave tile
// 32x64 for BOTH gate and up (16 f32 accumulators). KC=64 LDS double buffer.
// Grid: (DIM_HID/128, NTOK/128).
// ---------------------------------------------------------------------------
__global__ __launch_bounds__(256)
void gateup_kernel(const __bf16* __restrict__ xb,
                   const __bf16* __restrict__ wgb,
                   const __bf16* __restrict__ wub,
                   __bf16* __restrict__ act) {
  constexpr int A_OFF  = 0;
  constexpr int BG_OFF = 128 * RS;
  constexpr int BU_OFF = 256 * RS;
  constexpr int STAGE  = 384 * RS;                // 27648 elements
  __shared__ __bf16 smem[2 * STAGE];              // 110.6 KB (of 320 KB WGP LDS)

  const int tid  = threadIdx.x;
  const int wave = tid >> 5;
  const int lane = tid & 31;
  const int mw   = wave & 3;       // 4 M sub-tiles of 32 rows
  const int nw   = wave >> 2;      // 2 N sub-tiles of 64 cols
  const int r    = lane & 15;
  const int hi   = lane >> 4;
  const int m0   = blockIdx.y * 128;
  const int n0   = blockIdx.x * 128;

  v8f accg[2][4] = {};
  v8f accu[2][4] = {};

  const int NK = DIM_IN / KC;

#if !USE_TDM
  // fallback cooperative copy: tile 128 x 64 = 1024 v8bf chunks, 4/thread.
  v8bf rA[4], rG[4], rU[4];
#endif

#if USE_TDM
  if (tid < 32) {
    tdm_load_2d(xb  + (size_t)m0 * DIM_IN, smem + A_OFF,  128, DIM_IN);
    tdm_load_2d(wgb + (size_t)n0 * DIM_IN, smem + BG_OFF, 128, DIM_IN);
    tdm_load_2d(wub + (size_t)n0 * DIM_IN, smem + BU_OFF, 128, DIM_IN);
  }
#else
#pragma unroll
  for (int i = 0; i < 4; ++i) {
    const int c = tid + 256 * i, row = c >> 3, col = (c & 7) * 8;
    rA[i] = *(const v8bf*)(xb  + (size_t)(m0 + row) * DIM_IN + col);
    rG[i] = *(const v8bf*)(wgb + (size_t)(n0 + row) * DIM_IN + col);
    rU[i] = *(const v8bf*)(wub + (size_t)(n0 + row) * DIM_IN + col);
    *(v8bf*)(smem + A_OFF  + row * RS + col) = rA[i];
    *(v8bf*)(smem + BG_OFF + row * RS + col) = rG[i];
    *(v8bf*)(smem + BU_OFF + row * RS + col) = rU[i];
  }
  __syncthreads();
#endif

  for (int ks = 0; ks < NK; ++ks) {
    const int cur = (ks & 1) * STAGE;
    const int nxt = ((ks + 1) & 1) * STAGE;

#if USE_TDM
    if (tid < 32) {
      if (ks + 1 < NK) {
        const int k = (ks + 1) * KC;
        tdm_load_2d(xb  + (size_t)m0 * DIM_IN + k, smem + nxt + A_OFF,  128, DIM_IN);
        tdm_load_2d(wgb + (size_t)n0 * DIM_IN + k, smem + nxt + BG_OFF, 128, DIM_IN);
        tdm_load_2d(wub + (size_t)n0 * DIM_IN + k, smem + nxt + BU_OFF, 128, DIM_IN);
        __builtin_amdgcn_s_wait_tensorcnt(3);   // stage `cur` complete
      } else {
        __builtin_amdgcn_s_wait_tensorcnt(0);
      }
    }
    __syncthreads();          // stage `cur` visible to all waves
#else
    const bool havenext = (ks + 1 < NK);
    if (havenext) {
      const int k = (ks + 1) * KC;
#pragma unroll
      for (int i = 0; i < 4; ++i) {
        const int c = tid + 256 * i, row = c >> 3, col = (c & 7) * 8;
        rA[i] = *(const v8bf*)(xb  + (size_t)(m0 + row) * DIM_IN + k + col);
        rG[i] = *(const v8bf*)(wgb + (size_t)(n0 + row) * DIM_IN + k + col);
        rU[i] = *(const v8bf*)(wub + (size_t)(n0 + row) * DIM_IN + k + col);
      }
    }
#endif

    // ---- compute stage `cur` from LDS: 2 k-substeps x 16 WMMAs ----------
    {
      const __bf16* sA  = smem + cur + A_OFF  + (mw * 32 + r) * RS + hi * 16;
      const __bf16* sBg = smem + cur + BG_OFF + (nw * 64 + r) * RS + hi * 16;
      const __bf16* sBu = smem + cur + BU_OFF + (nw * 64 + r) * RS + hi * 16;
#pragma unroll
      for (int t = 0; t < 2; ++t) {
        const int ko = t * 32;
        v16bf a0  = load_frag_a(sA + ko);
        v16bf a1  = load_frag_a(sA + 16 * RS + ko);
        v16bf bg0 = load_frag_b(sBg + ko);
        v16bf bu0 = load_frag_b(sBu + ko);
        v16bf bg1 = load_frag_b(sBg + 16 * RS + ko);
        v16bf bu1 = load_frag_b(sBu + 16 * RS + ko);
        // j = 0 (set 0), then refill set 0 for j = 2
        accg[0][0] = wmma_bf16(a0, bg0, accg[0][0]);
        accg[1][0] = wmma_bf16(a1, bg0, accg[1][0]);
        accu[0][0] = wmma_bf16(a0, bu0, accu[0][0]);
        accu[1][0] = wmma_bf16(a1, bu0, accu[1][0]);
        bg0 = load_frag_b(sBg + 32 * RS + ko);
        bu0 = load_frag_b(sBu + 32 * RS + ko);
        // j = 1 (set 1), then refill set 1 for j = 3
        accg[0][1] = wmma_bf16(a0, bg1, accg[0][1]);
        accg[1][1] = wmma_bf16(a1, bg1, accg[1][1]);
        accu[0][1] = wmma_bf16(a0, bu1, accu[0][1]);
        accu[1][1] = wmma_bf16(a1, bu1, accu[1][1]);
        bg1 = load_frag_b(sBg + 48 * RS + ko);
        bu1 = load_frag_b(sBu + 48 * RS + ko);
        // j = 2 (set 0)
        accg[0][2] = wmma_bf16(a0, bg0, accg[0][2]);
        accg[1][2] = wmma_bf16(a1, bg0, accg[1][2]);
        accu[0][2] = wmma_bf16(a0, bu0, accu[0][2]);
        accu[1][2] = wmma_bf16(a1, bu0, accu[1][2]);
        // j = 3 (set 1)
        accg[0][3] = wmma_bf16(a0, bg1, accg[0][3]);
        accg[1][3] = wmma_bf16(a1, bg1, accg[1][3]);
        accu[0][3] = wmma_bf16(a0, bu1, accu[0][3]);
        accu[1][3] = wmma_bf16(a1, bu1, accu[1][3]);
      }
    }

#if USE_TDM
    __syncthreads();          // readers done before next TDM overwrites `nxt`
#else
    __syncthreads();
    if (havenext) {
#pragma unroll
      for (int i = 0; i < 4; ++i) {
        const int c = tid + 256 * i, row = c >> 3, col = (c & 7) * 8;
        *(v8bf*)(smem + nxt + A_OFF  + row * RS + col) = rA[i];
        *(v8bf*)(smem + nxt + BG_OFF + row * RS + col) = rG[i];
        *(v8bf*)(smem + nxt + BU_OFF + row * RS + col) = rU[i];
      }
    }
    __syncthreads();
#endif
  }

  // ---- SiLU(gate) * up epilogue, bf16 store -----------------------------
#pragma unroll
  for (int mi = 0; mi < 2; ++mi) {
    const int row_base = m0 + mw * 32 + mi * 16 + hi * 8;
#pragma unroll
    for (int j = 0; j < 4; ++j) {
      const int col = n0 + nw * 64 + j * 16 + r;
#pragma unroll
      for (int i = 0; i < 8; ++i) {
        float g = accg[mi][j][i];
        float u = accu[mi][j][i];
        float s = g * __builtin_amdgcn_rcpf(1.0f + __expf(-g));
        act[(size_t)(row_base + i) * DIM_HID + col] = (__bf16)(s * u);
      }
    }
  }
}

// ---------------------------------------------------------------------------
// Down GEMM: out[t, d] = sum_h act[t,h] * Wd[d,h]. K=8192, N=2048, f32 out.
// Block tile 128x128, wave tile 32x64. Grid: (DIM_IN/128, NTOK/128).
// ---------------------------------------------------------------------------
__global__ __launch_bounds__(256)
void down_kernel(const __bf16* __restrict__ actb,
                 const __bf16* __restrict__ wdb,
                 float* __restrict__ out) {
  constexpr int A_OFF = 0;
  constexpr int B_OFF = 128 * RS;
  constexpr int STAGE = 256 * RS;                 // 18432 elements
  __shared__ __bf16 smem[2 * STAGE];              // 73.7 KB

  const int tid  = threadIdx.x;
  const int wave = tid >> 5;
  const int lane = tid & 31;
  const int mw   = wave & 3;
  const int nw   = wave >> 2;
  const int r    = lane & 15;
  const int hi   = lane >> 4;
  const int m0   = blockIdx.y * 128;
  const int n0   = blockIdx.x * 128;

  v8f acc[2][4] = {};
  const int NK = DIM_HID / KC;

#if !USE_TDM
  v8bf rA[4], rB[4];
#endif

#if USE_TDM
  if (tid < 32) {
    tdm_load_2d(actb + (size_t)m0 * DIM_HID, smem + A_OFF, 128, DIM_HID);
    tdm_load_2d(wdb  + (size_t)n0 * DIM_HID, smem + B_OFF, 128, DIM_HID);
  }
#else
#pragma unroll
  for (int i = 0; i < 4; ++i) {
    const int c = tid + 256 * i, row = c >> 3, col = (c & 7) * 8;
    rA[i] = *(const v8bf*)(actb + (size_t)(m0 + row) * DIM_HID + col);
    rB[i] = *(const v8bf*)(wdb  + (size_t)(n0 + row) * DIM_HID + col);
    *(v8bf*)(smem + A_OFF + row * RS + col) = rA[i];
    *(v8bf*)(smem + B_OFF + row * RS + col) = rB[i];
  }
  __syncthreads();
#endif

  for (int ks = 0; ks < NK; ++ks) {
    const int cur = (ks & 1) * STAGE;
    const int nxt = ((ks + 1) & 1) * STAGE;

#if USE_TDM
    if (tid < 32) {
      if (ks + 1 < NK) {
        const int k = (ks + 1) * KC;
        tdm_load_2d(actb + (size_t)m0 * DIM_HID + k, smem + nxt + A_OFF, 128, DIM_HID);
        tdm_load_2d(wdb  + (size_t)n0 * DIM_HID + k, smem + nxt + B_OFF, 128, DIM_HID);
        __builtin_amdgcn_s_wait_tensorcnt(2);
      } else {
        __builtin_amdgcn_s_wait_tensorcnt(0);
      }
    }
    __syncthreads();
#else
    const bool havenext = (ks + 1 < NK);
    if (havenext) {
      const int k = (ks + 1) * KC;
#pragma unroll
      for (int i = 0; i < 4; ++i) {
        const int c = tid + 256 * i, row = c >> 3, col = (c & 7) * 8;
        rA[i] = *(const v8bf*)(actb + (size_t)(m0 + row) * DIM_HID + k + col);
        rB[i] = *(const v8bf*)(wdb  + (size_t)(n0 + row) * DIM_HID + k + col);
      }
    }
#endif

    {
      const __bf16* sA = smem + cur + A_OFF + (mw * 32 + r) * RS + hi * 16;
      const __bf16* sB = smem + cur + B_OFF + (nw * 64 + r) * RS + hi * 16;
#pragma unroll
      for (int t = 0; t < 2; ++t) {
        const int ko = t * 32;
        v16bf a0 = load_frag_a(sA + ko);
        v16bf a1 = load_frag_a(sA + 16 * RS + ko);
        v16bf b0 = load_frag_b(sB + ko);
        v16bf b1 = load_frag_b(sB + 16 * RS + ko);
        acc[0][0] = wmma_bf16(a0, b0, acc[0][0]);
        acc[1][0] = wmma_bf16(a1, b0, acc[1][0]);
        b0 = load_frag_b(sB + 32 * RS + ko);
        acc[0][1] = wmma_bf16(a0, b1, acc[0][1]);
        acc[1][1] = wmma_bf16(a1, b1, acc[1][1]);
        b1 = load_frag_b(sB + 48 * RS + ko);
        acc[0][2] = wmma_bf16(a0, b0, acc[0][2]);
        acc[1][2] = wmma_bf16(a1, b0, acc[1][2]);
        acc[0][3] = wmma_bf16(a0, b1, acc[0][3]);
        acc[1][3] = wmma_bf16(a1, b1, acc[1][3]);
      }
    }

#if USE_TDM
    __syncthreads();
#else
    __syncthreads();
    if (havenext) {
#pragma unroll
      for (int i = 0; i < 4; ++i) {
        const int c = tid + 256 * i, row = c >> 3, col = (c & 7) * 8;
        *(v8bf*)(smem + nxt + A_OFF + row * RS + col) = rA[i];
        *(v8bf*)(smem + nxt + B_OFF + row * RS + col) = rB[i];
      }
    }
    __syncthreads();
#endif
  }

#pragma unroll
  for (int mi = 0; mi < 2; ++mi) {
    const int row_base = m0 + mw * 32 + mi * 16 + hi * 8;
#pragma unroll
    for (int j = 0; j < 4; ++j) {
      const int col = n0 + nw * 64 + j * 16 + r;
#pragma unroll
      for (int i = 0; i < 8; ++i) {
        out[(size_t)(row_base + i) * DIM_IN + col] = acc[mi][j][i];
      }
    }
  }
}

// ---------------------------------------------------------------------------
extern "C" void kernel_launch(void* const* d_in, const int* in_sizes, int n_in,
                              void* d_out, int out_size, void* d_ws, size_t ws_size,
                              hipStream_t stream) {
  const float* x  = (const float*)d_in[0];
  const float* wg = (const float*)d_in[1];
  const float* wu = (const float*)d_in[2];
  const float* wd = (const float*)d_in[3];
  float* out = (float*)d_out;

  // Workspace layout (bf16 elements): xb | wgb | wub | wdb | act  (~268 MB)
  __bf16* base = (__bf16*)d_ws;
  size_t off = 0;
  __bf16* xb  = base + off; off += (size_t)NTOK    * DIM_IN;
  __bf16* wgb = base + off; off += (size_t)DIM_HID * DIM_IN;
  __bf16* wub = base + off; off += (size_t)DIM_HID * DIM_IN;
  __bf16* wdb = base + off; off += (size_t)DIM_IN  * DIM_HID;
  __bf16* act = base + off; off += (size_t)NTOK    * DIM_HID;

  const int cvt_block = 256;
  {
    int n4 = (NTOK * DIM_IN) / 4;
    cvt_f32_to_bf16<<<(n4 + cvt_block - 1) / cvt_block, cvt_block, 0, stream>>>(x, xb, n4);
  }
  {
    int n4 = (DIM_HID * DIM_IN) / 4;
    cvt_f32_to_bf16<<<(n4 + cvt_block - 1) / cvt_block, cvt_block, 0, stream>>>(wg, wgb, n4);
    cvt_f32_to_bf16<<<(n4 + cvt_block - 1) / cvt_block, cvt_block, 0, stream>>>(wu, wub, n4);
    cvt_f32_to_bf16<<<(n4 + cvt_block - 1) / cvt_block, cvt_block, 0, stream>>>(wd, wdb, n4);
  }

  dim3 gu_grid(DIM_HID / 128, NTOK / 128);
  gateup_kernel<<<gu_grid, 256, 0, stream>>>(xb, wgb, wub, act);

  dim3 dn_grid(DIM_IN / 128, NTOK / 128);
  down_kernel<<<dn_grid, 256, 0, stream>>>(act, wdb, out);
}
